// Seq2SeqModel_70257075028603
// MI455X (gfx1250) — matrix operand order repaired and checked by reading, not compiled, not run on hardware
//
#include <hip/hip_runtime.h>
#include <cmath>

// Sizes: T=2048, B=64, I=128, H=256, O=128
#define TT 2048
#define BB_ 64
#define II 128
#define HH 256
#define OO 128

typedef __bf16 bf16_t;
typedef __attribute__((ext_vector_type(16))) __bf16 v16bf;
typedef __attribute__((ext_vector_type(8)))  __bf16 v8bf;
typedef __attribute__((ext_vector_type(8)))  float  v8f;

#define WMMA_BF16(a, b, c) \
  __builtin_amdgcn_wmma_f32_16x16x32_bf16(false, (a), false, (b), (short)0, (c), false, false)

// --- optional CDNA5 async global->LDS path (guarded; falls back to reg prefetch)
#if __has_builtin(__builtin_amdgcn_global_load_async_to_lds_b128)
  #define HAVE_ASYNC_LDS 1
  // Signature (from clang diagnostic): param0 = AS(1) ptr to int vector_size(16),
  // param1 = LDS ptr (same payload type), param2/3 = imm offset / cpol.
  typedef int gv4i __attribute__((vector_size(16)));
  typedef __attribute__((address_space(1))) gv4i* gv4i_gptr;
  typedef __attribute__((address_space(3))) gv4i* gv4i_lptr;
  #if __has_builtin(__builtin_amdgcn_s_wait_asynccnt)
    #define WAIT_ASYNC() __builtin_amdgcn_s_wait_asynccnt(0)
  #else
    #define WAIT_ASYNC() asm volatile("s_wait_asynccnt 0x0" ::: "memory")
  #endif
#else
  #define HAVE_ASYNC_LDS 0
#endif

static __device__ __forceinline__ v16bf cat8(v8bf lo, v8bf hi) {
  return __builtin_shufflevector(lo, hi, 0,1,2,3,4,5,6,7,8,9,10,11,12,13,14,15);
}
static __device__ __forceinline__ v8f load8(const float* p) {
  float4 a = *(const float4*)p;
  float4 b = *(const float4*)(p + 4);
  v8f r; r[0]=a.x; r[1]=a.y; r[2]=a.z; r[3]=a.w; r[4]=b.x; r[5]=b.y; r[6]=b.z; r[7]=b.w;
  return r;
}
static __device__ __forceinline__ void store8(float* p, v8f v) {
  float4 a = {v[0], v[1], v[2], v[3]};
  float4 b = {v[4], v[5], v[6], v[7]};
  *(float4*)p = a; *(float4*)(p + 4) = b;
}
static __device__ __forceinline__ void store8bf(bf16_t* p, v8f v) {
  v8bf o;
#pragma unroll
  for (int e = 0; e < 8; ++e) o[e] = (bf16_t)v[e];
  *(v8bf*)p = o;
}

// ---------------------------------------------------------------------------
// Projection GEMM: Cout[M x N] = A[M x K] * W[N x K]^T (+ bias)
// Block = 256 threads (8 waves), tile 64 rows x 64 cols, k-step 32.
// ---------------------------------------------------------------------------
template <int K, bool BIAS>
__global__ __launch_bounds__(256)
void proj_kernel(const float* __restrict__ A, const float* __restrict__ W,
                 const float* __restrict__ bias, float* __restrict__ Cout, int N)
{
  __shared__ bf16_t As[64][40];   // padded stride 40 -> conflict-free frag reads
  __shared__ bf16_t Bs[64][40];

  const int tid  = threadIdx.x;
  const int lane = tid & 31, wave = tid >> 5;
  const int hi   = lane >> 4, lm = lane & 15;
  const int rTile = wave >> 1;           // 0..3
  const int cT0   = (wave & 1) * 2;      // 0 or 2 (two col tiles per wave)
  const size_t rowBase = (size_t)blockIdx.x * 64;
  const int    colBase = blockIdx.y * 64;
  const int srow = tid >> 2, seg = tid & 3;   // staging: 64 rows x 4 segs of 8 floats

  v8f c0 = {}, c1 = {};

  for (int k0 = 0; k0 < K; k0 += 32) {
    const float* ap = A + (rowBase + srow) * K + k0 + seg * 8;
    float4 a0 = *(const float4*)ap;
    float4 a1 = *(const float4*)(ap + 4);
    const float* wp = W + (size_t)(colBase + srow) * K + k0 + seg * 8;
    float4 w0 = *(const float4*)wp;
    float4 w1 = *(const float4*)(wp + 4);
    __syncthreads();
    v8bf av = { (bf16_t)a0.x, (bf16_t)a0.y, (bf16_t)a0.z, (bf16_t)a0.w,
                (bf16_t)a1.x, (bf16_t)a1.y, (bf16_t)a1.z, (bf16_t)a1.w };
    v8bf wv = { (bf16_t)w0.x, (bf16_t)w0.y, (bf16_t)w0.z, (bf16_t)w0.w,
                (bf16_t)w1.x, (bf16_t)w1.y, (bf16_t)w1.z, (bf16_t)w1.w };
    *(v8bf*)&As[srow][seg * 8] = av;
    *(v8bf*)&Bs[srow][seg * 8] = wv;
    __syncthreads();

    const bf16_t* ar = &As[rTile * 16 + lm][0];
    v16bf af = cat8(*(const v8bf*)(ar + hi * 8), *(const v8bf*)(ar + 16 + hi * 8));
    const bf16_t* b0r = &Bs[cT0 * 16 + lm][0];
    v16bf bf0 = cat8(*(const v8bf*)(b0r + hi * 8), *(const v8bf*)(b0r + 16 + hi * 8));
    const bf16_t* b1r = &Bs[cT0 * 16 + 16 + lm][0];
    v16bf bf1 = cat8(*(const v8bf*)(b1r + hi * 8), *(const v8bf*)(b1r + 16 + hi * 8));

    c0 = WMMA_BF16(af, bf0, c0);
    c1 = WMMA_BF16(af, bf1, c1);
  }

  float bias0 = 0.f, bias1 = 0.f;
  if (BIAS) {
    bias0 = bias[colBase + cT0 * 16 + lm];
    bias1 = bias[colBase + cT0 * 16 + 16 + lm];
  }
  const int ncol = colBase + cT0 * 16 + lm;
#pragma unroll
  for (int r = 0; r < 8; ++r) {           // C/D: M = r + 8*hi, N = lane&15
    size_t m = rowBase + rTile * 16 + r + 8 * hi;
    Cout[m * (size_t)N + ncol]      = c0[r] + bias0;
    Cout[m * (size_t)N + ncol + 16] = c1[r] + bias1;
  }
}

// ---------------------------------------------------------------------------
// Persistent single-WGP recurrent scan, 512 threads = 16 wave32 waves.
// hid[t][b][n] initially holds xin + b_hh; overwritten in place with h_t.
// Wave mapping: bG = wave&1 -> b-tiles {bG*32, bG*32+16};
//               nG = wave>>1 -> n-tiles {nG*32, nG*32+16}.
// Each wave: 4 C tiles, W_hh register-resident = 2x8 v16bf = 128 VGPRs/lane.
// xin[t+1] prefetch: async global->LDS (ASYNCcnt) when available.
// ---------------------------------------------------------------------------
__global__ __launch_bounds__(512)
void scan_kernel(const float* __restrict__ W_hh, float* __restrict__ hid)
{
  extern __shared__ char smem[];
  float*  xin_s = (float*)smem;              // [2][64*260] fp32 (padded stride 260)
  float*  hf    = xin_s + 2 * 64 * 260;      // [64*260]  fp32 h staging
  bf16_t* hb    = (bf16_t*)(hf + 64 * 260);  // [2][64*264] bf16 h (padded stride 264)

  const int tid  = threadIdx.x;
  const int lane = tid & 31, wave = tid >> 5;
  const int hi   = lane >> 4, lm = lane & 15;
  const int bG = wave & 1;
  const int n0 = (wave >> 1) * 32;           // n-tiles: n0, n0+16
  const int b0 = bG * 32;                    // b-tiles: b0, b0+16
  const int cb = tid >> 3, coff = (tid & 7) * 32;  // 512 thr x 32 floats global copy

  // --- persistent W_hh A-fragments (bf16), 128 VGPRs/lane ---
  v16bf wA0[8], wA1[8];
#pragma unroll
  for (int s = 0; s < 8; ++s) {
    const float* w0 = W_hh + (size_t)(n0 + lm) * HH + s * 32 + hi * 8;
    const float* w1 = W_hh + (size_t)(n0 + 16 + lm) * HH + s * 32 + hi * 8;
    v16bf f0, f1;
#pragma unroll
    for (int e = 0; e < 8; ++e) {
      f0[e] = (bf16_t)w0[e]; f0[e + 8] = (bf16_t)w0[16 + e];
      f1[e] = (bf16_t)w1[e]; f1[e + 8] = (bf16_t)w1[16 + e];
    }
    wA0[s] = f0; wA1[s] = f1;
  }

  // --- init: h0 = 0 (bf16 buffer 0), xin_s[0] <- hid[0] ---
  for (int idx = tid; idx < 64 * 264; idx += 512) hb[idx] = (bf16_t)0.0f;
#pragma unroll
  for (int j = 0; j < 8; ++j)
    *(float4*)(xin_s + cb * 260 + coff + j * 4) =
        *(const float4*)(hid + (size_t)cb * HH + coff + j * 4);
  __syncthreads();

  for (int t = 0; t < TT; ++t) {
    const int cur = t & 1, nxt = cur ^ 1;

    // (a) coalesced store of h_{t-1} to global hiddens
    if (t > 0) {
#pragma unroll
      for (int j = 0; j < 8; ++j)
        *(float4*)(hid + (size_t)(t - 1) * (BB_ * HH) + cb * HH + coff + j * 4) =
            *(const float4*)(hf + cb * 260 + coff + j * 4);
    }

    // (b) prefetch xin[t+1] into xin_s[nxt] (idle buffer; all reads of it
    //     finished before step t-1's first barrier)
#if HAVE_ASYNC_LDS
    if (t < TT - 1) {
      float* gsrc = hid + (size_t)(t + 1) * (BB_ * HH) + cb * HH + coff;
      float* ldst = xin_s + nxt * (64 * 260) + cb * 260 + coff;
#pragma unroll
      for (int j = 0; j < 8; ++j)
        __builtin_amdgcn_global_load_async_to_lds_b128(
            (gv4i_gptr)(gsrc + j * 4), (gv4i_lptr)(ldst + j * 4), 0, 0);
    }
#else
    float4 pf[8];
    if (t < TT - 1) {
#pragma unroll
      for (int j = 0; j < 8; ++j)
        pf[j] = *(const float4*)(hid + (size_t)(t + 1) * (BB_ * HH) + cb * HH + coff + j * 4);
    }
#endif

    // (c) C[n][b] = xin'[b][n] + sum_k W_hh[n][k] * h[b][k]   (b_hh pre-added)
    const float* xr0 = xin_s + cur * (64 * 260) + (b0 + lm) * 260;
    const float* xr1 = xr0 + 16 * 260;
    v8f c00 = load8(xr0 + n0 + 8 * hi);
    v8f c10 = load8(xr0 + n0 + 16 + 8 * hi);
    v8f c01 = load8(xr1 + n0 + 8 * hi);
    v8f c11 = load8(xr1 + n0 + 16 + 8 * hi);

    const bf16_t* hr0 = hb + cur * (64 * 264) + (b0 + lm) * 264;
    const bf16_t* hr1 = hr0 + 16 * 264;
#pragma unroll
    for (int s = 0; s < 8; ++s) {
      v16bf bh0 = cat8(*(const v8bf*)(hr0 + s * 32 + hi * 8),
                       *(const v8bf*)(hr0 + s * 32 + 16 + hi * 8));
      v16bf bh1 = cat8(*(const v8bf*)(hr1 + s * 32 + hi * 8),
                       *(const v8bf*)(hr1 + s * 32 + 16 + hi * 8));
      c00 = WMMA_BF16(wA0[s], bh0, c00);
      c10 = WMMA_BF16(wA1[s], bh0, c10);
      c01 = WMMA_BF16(wA0[s], bh1, c01);
      c11 = WMMA_BF16(wA1[s], bh1, c11);
    }
#pragma unroll
    for (int r = 0; r < 8; ++r) {
      c00[r] = tanhf(c00[r]); c10[r] = tanhf(c10[r]);
      c01[r] = tanhf(c01[r]); c11[r] = tanhf(c11[r]);
    }

    __syncthreads();  // (d) all reads of hf / hb[cur] / xin_s[cur] complete

    // (e) publish h_t (vectorized, conflict-free via padded strides)
    float* hf0 = hf + (b0 + lm) * 260;
    float* hf1 = hf0 + 16 * 260;
    store8(hf0 + n0 + 8 * hi, c00); store8(hf0 + n0 + 16 + 8 * hi, c10);
    store8(hf1 + n0 + 8 * hi, c01); store8(hf1 + n0 + 16 + 8 * hi, c11);
    bf16_t* hb0 = hb + nxt * (64 * 264) + (b0 + lm) * 264;
    bf16_t* hb1 = hb0 + 16 * 264;
    store8bf(hb0 + n0 + 8 * hi, c00); store8bf(hb0 + n0 + 16 + 8 * hi, c10);
    store8bf(hb1 + n0 + 8 * hi, c01); store8bf(hb1 + n0 + 16 + 8 * hi, c11);

#if HAVE_ASYNC_LDS
    if (t < TT - 1) WAIT_ASYNC();  // own async LDS writes done before barrier
#else
    if (t < TT - 1) {
#pragma unroll
      for (int j = 0; j < 8; ++j)
        *(float4*)(xin_s + nxt * (64 * 260) + cb * 260 + coff + j * 4) = pf[j];
    }
#endif
    __syncthreads();  // (f) writes visible before next step
  }

  // final h_{T-1}
#pragma unroll
  for (int j = 0; j < 8; ++j)
    *(float4*)(hid + (size_t)(TT - 1) * (BB_ * HH) + cb * HH + coff + j * 4) =
        *(const float4*)(hf + cb * 260 + coff + j * 4);
}

// ---------------------------------------------------------------------------
extern "C" void kernel_launch(void* const* d_in, const int* in_sizes, int n_in,
                              void* d_out, int out_size, void* d_ws, size_t ws_size,
                              hipStream_t stream)
{
  (void)in_sizes; (void)n_in; (void)out_size; (void)d_ws; (void)ws_size;
  const float* x     = (const float*)d_in[0];
  const float* W_in  = (const float*)d_in[1];
  const float* W_hh  = (const float*)d_in[2];
  const float* b_hh  = (const float*)d_in[3];
  const float* W_out = (const float*)d_in[4];
  const float* b_out = (const float*)d_in[5];

  float* hid  = (float*)d_out;                          // [T*B*H] hiddens (xin in-place)
  float* outp = hid + (size_t)TT * BB_ * HH;            // [T*B*O] output

  // 1) xin = x @ W_in^T + b_hh  (bias pre-added so the scan skips it)
  proj_kernel<II, true><<<dim3((TT * BB_) / 64, HH / 64), 256, 0, stream>>>(
      x, W_in, b_hh, hid, HH);

  // 2) recurrent scan in place (single persistent WGP, ~261KB dynamic LDS)
  size_t smem = (size_t)(3 * 64 * 260) * sizeof(float) + (size_t)(2 * 64 * 264) * sizeof(bf16_t);
  scan_kernel<<<dim3(1), 512, smem, stream>>>(W_hh, hid);

  // 3) output = hiddens @ W_out^T + b_out
  proj_kernel<HH, true><<<dim3((TT * BB_) / 64, OO / 64), 256, 0, stream>>>(
      hid, W_out, b_out, outp, OO);
}
